// FourierNeuralOperator_90099823935689
// MI455X (gfx1250) — matrix operand order repaired and checked by reading, not compiled, not run on hardware
//
#include <hip/hip_runtime.h>
#include <hip/hip_bf16.h>
#include <math.h>

// FNO dims (match reference)
#define L_LAYERS 8
#define H_DIM    512
#define M_MODES  32
#define KB_BITS  256
#define B_BATCH  8
#define S_LEN    2048
#define EPS_LN   1e-5f

typedef __attribute__((ext_vector_type(16))) __bf16 v16bf;
typedef __attribute__((ext_vector_type(8)))  float  v8f;
typedef unsigned int u32x4 __attribute__((ext_vector_type(4)));
typedef int          i32x8 __attribute__((ext_vector_type(8)));
typedef int          i32x4 __attribute__((ext_vector_type(4)));

#define LDF 36   // f32 LDS tile leading dim (32 + 4 pad) == TDM pad 32dw+4dw

__device__ __forceinline__ v8f vzero8() {
    v8f z = {0.f, 0.f, 0.f, 0.f, 0.f, 0.f, 0.f, 0.f};
    return z;
}

__device__ __forceinline__ v8f wmma_bf16(v16bf a, v16bf b, v8f c) {
    return __builtin_amdgcn_wmma_f32_16x16x32_bf16(
        false, a, false, b, (short)0, c, false, false);
}

__device__ __forceinline__ float gelu_f(float x) {
    return 0.5f * x * (1.f + erff(x * 0.70710678118654752f));
}

__device__ __forceinline__ unsigned lds_off(const void* p) {
    // generic shared-aperture address: low 32 bits are the LDS byte offset
    return (unsigned)(size_t)p;
}

// ---- CDNA5 Tensor Data Mover: one 2D f32 tile -> padded LDS rows -----------
// D# per ISA 8.3/8.4: count=1, type=2 (image), data_size=2 (4B),
// pad_enable=1, pad_interval=4 (32 DWORDs), pad_amount=3 (4 DWORDs)
// -> LDS rows land at 36-float stride (LDF), matching the fragment loaders.
__device__ __forceinline__ void tdm_load_tile_2d(const float* gtile, unsigned ldsaddr,
                                                 unsigned rows, unsigned cols,
                                                 unsigned row_stride_elems) {
    unsigned long long ga = (unsigned long long)(size_t)gtile;
    u32x4 g0;
    g0[0] = 1u;                                            // count=1 valid descriptor
    g0[1] = ldsaddr;                                       // lds_addr (bytes)
    g0[2] = (unsigned)(ga & 0xFFFFFFFFu);                  // global_addr[31:0]
    g0[3] = ((unsigned)(ga >> 32) & 0x01FFFFFFu) | 0x80000000u; // addr[56:32] | type=2
    i32x8 g1;
    g1[0] = (int)((2u << 16) | (1u << 20) | (4u << 22) | (3u << 25));
    g1[1] = (int)(cols << 16);                             // tensor_dim0[15:0]
    g1[2] = (int)((cols >> 16) | (rows << 16));            // tensor_dim0 hi | tensor_dim1 lo
    g1[3] = (int)((rows >> 16) | (cols << 16));            // tensor_dim1 hi | tile_dim0
    g1[4] = (int)rows;                                     // tile_dim1 (tile_dim2=0)
    g1[5] = (int)row_stride_elems;                         // tensor_dim0_stride[31:0]
    g1[6] = 0;
    g1[7] = 0;
    i32x4 z4 = {0, 0, 0, 0};
#if defined(__clang_major__) && (__clang_major__ >= 23)
    i32x8 z8 = {0, 0, 0, 0, 0, 0, 0, 0};
    __builtin_amdgcn_tensor_load_to_lds(g0, g1, z4, z4, z8, 0);
#else
    __builtin_amdgcn_tensor_load_to_lds(g0, g1, z4, z4, 0);
#endif
}
template <int N>
__device__ __forceinline__ void wait_tensor() {
    __builtin_amdgcn_s_wait_tensorcnt(N);   // immediate required
}

// ---- CDNA5 async global->LDS staging (ASYNCcnt-tracked) --------------------
__device__ __forceinline__ void async_tile_row(const float* g, float* l) {
    const unsigned lo = lds_off(l);
#pragma unroll
    for (int c = 0; c < 8; ++c) {
        asm volatile("global_load_async_to_lds_b128 %0, %1, off offset:%2"
                     :: "v"(lo), "v"(g), "i"(c * 16) : "memory");
    }
}
__device__ __forceinline__ void async_half_row(const float* g, float* l) {
    const unsigned lo = lds_off(l);
#pragma unroll
    for (int c = 0; c < 4; ++c) {
        asm volatile("global_load_async_to_lds_b128 %0, %1, off offset:%2"
                     :: "v"(lo), "v"(g), "i"(c * 16) : "memory");
    }
}
__device__ __forceinline__ void wait_async0() {
    asm volatile("s_wait_asynccnt 0x0" ::: "memory");
}
__device__ __forceinline__ void wait_ds0() {
    asm volatile("s_wait_dscnt 0x0" ::: "memory");
}

// ---- fragment assembly from f32 LDS tiles ----------------------------------
__device__ __forceinline__ void cvt4(v16bf& d, int base, float4 f) {
    d[base + 0] = (__bf16)f.x; d[base + 1] = (__bf16)f.y;
    d[base + 2] = (__bf16)f.z; d[base + 3] = (__bf16)f.w;
}

// A fragment (ISA 7.12.2): lane L<16 -> row M=L, K={kb..kb+7, 16+kb..16+kb+7}.
__device__ __forceinline__ void frag_a_f32(const float* T, int row0, v16bf& a) {
    const int lane = threadIdx.x & 31;
    const int r  = row0 + (lane & 15);
    const int kb = (lane >> 4) << 3;
    const float4* p = (const float4*)&T[r * LDF + kb];
    const float4* q = (const float4*)&T[r * LDF + 16 + kb];
    cvt4(a, 0, p[0]); cvt4(a, 4, p[1]);
    cvt4(a, 8, q[0]); cvt4(a, 12, q[1]);
}

// B fragment from a [n][k] K-contiguous tile -> 4 x ds_load_b128.
__device__ __forceinline__ void frag_b_f32_rowK(const float* T, int col0, v16bf& b) {
    const int lane = threadIdx.x & 31;
    const int c  = col0 + (lane & 15);
    const int k0 = (lane >> 4) << 3;
    const float4* p = (const float4*)&T[c * LDF + k0];
    const float4* q = (const float4*)&T[c * LDF + 16 + k0];
    cvt4(b, 0, p[0]); cvt4(b, 4, p[1]); cvt4(b, 8, q[0]); cvt4(b, 12, q[1]);
}

// B fragment from a [k][n] tile (true transpose): strided b32 gathers.
__device__ __forceinline__ void frag_b_f32_colK(const float* T, int col0, v16bf& b) {
    const int lane = threadIdx.x & 31;
    const int c  = col0 + (lane & 15);
    const int k0 = (lane >> 4) << 3;
#pragma unroll
    for (int i = 0; i < 16; ++i) b[i] = (__bf16)T[(k0 + i + ((i >> 3) << 3)) * LDF + c];
}

// ---------------------------------------------------------------------------
// Generic C[M,N] = act(A[M,K] @ W[N,K]^T + bias). One wave -> 32x32 tile.
// Full tiles: double-buffered TDM staging (TENSORcnt overlap with WMMA).
// ACT: 0 none, 1 gelu, 2 sigmoid
// ---------------------------------------------------------------------------
template <int ACT>
__global__ __launch_bounds__(32)
void gemm_aw_kernel(const float* __restrict__ A, const float* __restrict__ W,
                    const float* __restrict__ bias, float* __restrict__ C,
                    int M, int N, int K) {
    __shared__ __align__(16) float Asf[2][32 * LDF];
    __shared__ __align__(16) float Bsf[2][32 * LDF];
    const int m0 = blockIdx.x * 32;
    const int n0 = blockIdx.y * 32;
    const int lane = threadIdx.x;
    v8f acc[2][2];
    acc[0][0] = vzero8(); acc[0][1] = vzero8();
    acc[1][0] = vzero8(); acc[1][1] = vzero8();

    const bool full = (m0 + 32 <= M);      // N tiles are always exact here
    const int am = m0 + lane;
    const float* Abase = A + (size_t)m0 * K;
    const float* Wbase = W + (size_t)n0 * K;

    if (full) {
        // prologue: stage k-step 0 into buffer 0
        tdm_load_tile_2d(Abase, lds_off(Asf[0]), 32, 32, (unsigned)K);
        tdm_load_tile_2d(Wbase, lds_off(Bsf[0]), 32, 32, (unsigned)K);
    }

    for (int k0 = 0; k0 < K; k0 += 32) {
        const int cur = (k0 >> 5) & 1;
        if (full) {
            if (k0 + 32 < K) {
                wait_ds0();   // old reads of the buffer we are about to refill
                tdm_load_tile_2d(Abase + k0 + 32, lds_off(Asf[cur ^ 1]), 32, 32, (unsigned)K);
                tdm_load_tile_2d(Wbase + k0 + 32, lds_off(Bsf[cur ^ 1]), 32, 32, (unsigned)K);
                wait_tensor<2>();   // current buffer's two tile loads complete
            } else {
                wait_tensor<0>();
            }
        } else {
            wait_ds0();
            if (am < M) {
                const float* ap = A + (size_t)am * K + k0;
#pragma unroll
                for (int j = 0; j < 32; ++j) Asf[cur][lane * LDF + j] = ap[j];
            } else {
#pragma unroll
                for (int j = 0; j < 32; ++j) Asf[cur][lane * LDF + j] = 0.f;
            }
            const float* wp = W + (size_t)(n0 + lane) * K + k0;
#pragma unroll
            for (int j = 0; j < 32; ++j) Bsf[cur][lane * LDF + j] = wp[j];
        }
        __syncthreads();
        v16bf a0, a1, b0, b1;
        frag_a_f32(Asf[cur], 0, a0);
        frag_a_f32(Asf[cur], 16, a1);
        frag_b_f32_rowK(Bsf[cur], 0, b0);
        frag_b_f32_rowK(Bsf[cur], 16, b1);
        acc[0][0] = wmma_bf16(a0, b0, acc[0][0]);
        acc[0][1] = wmma_bf16(a0, b1, acc[0][1]);
        acc[1][0] = wmma_bf16(a1, b0, acc[1][0]);
        acc[1][1] = wmma_bf16(a1, b1, acc[1][1]);
        __syncthreads();
    }

    const int col = lane & 15;
    const int rb  = (lane >> 4) << 3;
#pragma unroll
    for (int tn = 0; tn < 2; ++tn) {
        const int n = n0 + tn * 16 + col;
        if (n >= N) continue;
        const float bv = bias ? bias[n] : 0.f;
#pragma unroll
        for (int tm = 0; tm < 2; ++tm) {
#pragma unroll
            for (int r = 0; r < 8; ++r) {
                const int m = m0 + tm * 16 + rb + r;
                if (m < M) {
                    float v = acc[tm][tn][r] + bv;
                    if (ACT == 1)      v = gelu_f(v);
                    else if (ACT == 2) v = 1.f / (1.f + __expf(-v));
                    C[(size_t)m * N + n] = v;
                }
            }
        }
    }
}

// ---------------------------------------------------------------------------
// Truncated forward DFT: spec[b][0..31] = Re, spec[b][32..63] = Im.
// A = F[64 x S] analytic (registers), B = h[b] [S x H] async-staged.
// ---------------------------------------------------------------------------
__global__ __launch_bounds__(32)
void fno_fwd_dft_kernel(const float* __restrict__ h, float* __restrict__ spec) {
    __shared__ __align__(16) float Bsf[32 * LDF];
    const int m0 = blockIdx.x * 32;            // 0 or 32
    const int n0 = blockIdx.y * 32;
    const int b  = blockIdx.z;
    const int lane = threadIdx.x;
    const float w = 6.2831853071795864f / (float)S_LEN;
    v8f acc[2][2];
    acc[0][0] = vzero8(); acc[0][1] = vzero8();
    acc[1][0] = vzero8(); acc[1][1] = vzero8();

    for (int k0 = 0; k0 < S_LEN; k0 += 32) {
        wait_ds0();
        async_tile_row(h + ((size_t)b * S_LEN + k0 + lane) * H_DIM + n0,
                       Bsf + lane * LDF);
        wait_async0();
        __syncthreads();

        v16bf a0, a1, b0, b1;
        const int r  = lane & 15;
        const int kb = (lane >> 4) << 3;
#pragma unroll
        for (int t = 0; t < 2; ++t) {
            const int mrow = m0 + t * 16 + r;
            v16bf& a = t ? a1 : a0;
#pragma unroll
            for (int j = 0; j < 8; ++j) {
                const int s1 = k0 + kb + j;
                const int s2 = k0 + 16 + kb + j;
                float v1, v2;
                if (mrow < 32) {
                    v1 = __cosf(w * (float)(mrow * s1));
                    v2 = __cosf(w * (float)(mrow * s2));
                } else {
                    v1 = -__sinf(w * (float)((mrow - 32) * s1));
                    v2 = -__sinf(w * (float)((mrow - 32) * s2));
                }
                a[j]     = (__bf16)v1;
                a[8 + j] = (__bf16)v2;
            }
        }
        frag_b_f32_colK(Bsf, 0, b0);
        frag_b_f32_colK(Bsf, 16, b1);
        acc[0][0] = wmma_bf16(a0, b0, acc[0][0]);
        acc[0][1] = wmma_bf16(a0, b1, acc[0][1]);
        acc[1][0] = wmma_bf16(a1, b0, acc[1][0]);
        acc[1][1] = wmma_bf16(a1, b1, acc[1][1]);
        __syncthreads();
    }

    const int col = lane & 15;
    const int rb  = (lane >> 4) << 3;
#pragma unroll
    for (int tn = 0; tn < 2; ++tn) {
        const int n = n0 + tn * 16 + col;
#pragma unroll
        for (int tm = 0; tm < 2; ++tm) {
#pragma unroll
            for (int r = 0; r < 8; ++r) {
                const int m = m0 + tm * 16 + rb + r;
                spec[((size_t)b * 64 + m) * H_DIM + n] = acc[tm][tn][r];
            }
        }
    }
}

// ---------------------------------------------------------------------------
// Per-mode complex GEMM: out[b,m,:] = spec[b,m,:] @ (Wr + i Wi)[m].
// A tile 16x32 async-staged; Wr/Wi tiles TDM-staged.
// P = A@Wr, Q = A@Wi; out_r = P_top - Q_bot, out_i = Q_top + P_bot (shfl_xor16)
// ---------------------------------------------------------------------------
__global__ __launch_bounds__(32)
void fno_spectral_kernel(const float* __restrict__ spec,
                         const float* __restrict__ Wr,
                         const float* __restrict__ Wi,
                         float* __restrict__ specout) {
    __shared__ __align__(16) float Asf[16 * LDF];
    __shared__ __align__(16) float Brf[32 * LDF];
    __shared__ __align__(16) float Bif[32 * LDF];
    const int m  = blockIdx.x;             // mode
    const int n0 = blockIdx.y * 32;
    const int lane = threadIdx.x;
    v8f P0 = vzero8(), P1 = vzero8(), Q0 = vzero8(), Q1 = vzero8();

    const int arow  = lane >> 1;           // 0..15
    const int ahalf = lane & 1;
    const int abb   = arow & 7;
    const int amrow = (arow < 8) ? m : (32 + m);

    for (int k0 = 0; k0 < H_DIM; k0 += 32) {
        wait_ds0();
        tdm_load_tile_2d(Wr + ((size_t)m * H_DIM + k0) * H_DIM + n0,
                         lds_off(Brf), 32, 32, H_DIM);
        tdm_load_tile_2d(Wi + ((size_t)m * H_DIM + k0) * H_DIM + n0,
                         lds_off(Bif), 32, 32, H_DIM);
        async_half_row(spec + ((size_t)abb * 64 + amrow) * H_DIM + k0 + ahalf * 16,
                       Asf + arow * LDF + ahalf * 16);
        wait_async0();
        wait_tensor<0>();
        __syncthreads();
        v16bf a, br0, br1, bi0, bi1;
        frag_a_f32(Asf, 0, a);
        frag_b_f32_colK(Brf, 0, br0);
        frag_b_f32_colK(Brf, 16, br1);
        frag_b_f32_colK(Bif, 0, bi0);
        frag_b_f32_colK(Bif, 16, bi1);
        P0 = wmma_bf16(a, br0, P0);
        P1 = wmma_bf16(a, br1, P1);
        Q0 = wmma_bf16(a, bi0, Q0);
        Q1 = wmma_bf16(a, bi1, Q1);
        __syncthreads();
    }

    const int  col = lane & 15;
    const bool top = lane < 16;
#pragma unroll
    for (int t = 0; t < 2; ++t) {
        v8f& P = t ? P1 : P0;
        v8f& Q = t ? Q1 : Q0;
#pragma unroll
        for (int r = 0; r < 8; ++r) {
            float q = Q[r];
            float other = __shfl_xor(q, 16, 32);
            float val = top ? (P[r] - other) : (P[r] + other);
            const int mrow = top ? m : (32 + m);
            const int n = n0 + t * 16 + col;
            specout[((size_t)r * 64 + mrow) * H_DIM + n] = val;
        }
    }
}

// ---------------------------------------------------------------------------
// Truncated inverse DFT (irfft of 32-mode spectrum), B tile TDM-staged.
// ---------------------------------------------------------------------------
__global__ __launch_bounds__(32)
void fno_inv_dft_kernel(const float* __restrict__ specout, float* __restrict__ fb) {
    __shared__ __align__(16) float Bsf[32 * LDF];
    const int s0 = blockIdx.x * 32;
    const int n0 = blockIdx.y * 32;
    const int b  = blockIdx.z;
    const int lane = threadIdx.x;
    const float w   = 6.2831853071795864f / (float)S_LEN;
    const float inv = 1.f / (float)S_LEN;
    v8f acc[2][2];
    acc[0][0] = vzero8(); acc[0][1] = vzero8();
    acc[1][0] = vzero8(); acc[1][1] = vzero8();

    for (int k0 = 0; k0 < 64; k0 += 32) {
        wait_ds0();
        tdm_load_tile_2d(specout + ((size_t)b * 64 + k0) * H_DIM + n0,
                         lds_off(Bsf), 32, 32, H_DIM);
        wait_tensor<0>();
        __syncthreads();

        v16bf a0, a1, b0, b1;
        const int r  = lane & 15;
        const int kb = (lane >> 4) << 3;
#pragma unroll
        for (int t = 0; t < 2; ++t) {
            const int s = s0 + t * 16 + r;
            v16bf& a = t ? a1 : a0;
#pragma unroll
            for (int j = 0; j < 8; ++j) {
#pragma unroll
                for (int half = 0; half < 2; ++half) {
                    const int mp = k0 + half * 16 + kb + j;
                    float v;
                    if (mp < 32) {
                        const float cm = (mp == 0) ? 1.f : 2.f;
                        v = cm * inv * __cosf(w * (float)(mp * s));
                    } else {
                        v = -2.f * inv * __sinf(w * (float)((mp - 32) * s));
                    }
                    a[half * 8 + j] = (__bf16)v;
                }
            }
        }
        frag_b_f32_colK(Bsf, 0, b0);
        frag_b_f32_colK(Bsf, 16, b1);
        acc[0][0] = wmma_bf16(a0, b0, acc[0][0]);
        acc[0][1] = wmma_bf16(a0, b1, acc[0][1]);
        acc[1][0] = wmma_bf16(a1, b0, acc[1][0]);
        acc[1][1] = wmma_bf16(a1, b1, acc[1][1]);
        __syncthreads();
    }

    const int col = lane & 15;
    const int rb  = (lane >> 4) << 3;
#pragma unroll
    for (int tn = 0; tn < 2; ++tn) {
        const int n = n0 + tn * 16 + col;
#pragma unroll
        for (int tm = 0; tm < 2; ++tm) {
#pragma unroll
            for (int r = 0; r < 8; ++r) {
                const int s = s0 + tm * 16 + rb + r;
                fb[((size_t)b * S_LEN + s) * H_DIM + n] = acc[tm][tn][r];
            }
        }
    }
}

// ---------------------------------------------------------------------------
// Elementwise / reduction kernels
// ---------------------------------------------------------------------------
__global__ void input_proj_kernel(const float* __restrict__ x,
                                  const float* __restrict__ in_w,
                                  const float* __restrict__ in_b,
                                  float* __restrict__ h) {
    const int i = blockIdx.x * blockDim.x + threadIdx.x;
    const int total = B_BATCH * S_LEN * H_DIM;
    if (i < total) {
        const int c   = i & (H_DIM - 1);
        const int row = i >> 9;
        h[i] = x[row] * in_w[c] + in_b[c];
    }
}

__global__ __launch_bounds__(128)
void combine_ln_kernel(const float* __restrict__ fb, const float* __restrict__ cv,
                       const float* __restrict__ g, const float* __restrict__ bta,
                       float* __restrict__ h) {
    const int row = blockIdx.x;
    const size_t base = (size_t)row * H_DIM;
    const int tid = threadIdx.x;
    float y[4];
    float s = 0.f, s2 = 0.f;
#pragma unroll
    for (int i = 0; i < 4; ++i) {
        const int c = tid + i * 128;
        const float v = fb[base + c] + cv[base + c];
        y[i] = v;
        s += v; s2 += v * v;
    }
#pragma unroll
    for (int off = 16; off > 0; off >>= 1) {
        s  += __shfl_xor(s, off, 32);
        s2 += __shfl_xor(s2, off, 32);
    }
    __shared__ float sh[8];
    const int wid = tid >> 5;
    if ((tid & 31) == 0) { sh[wid] = s; sh[4 + wid] = s2; }
    __syncthreads();
    s  = sh[0] + sh[1] + sh[2] + sh[3];
    s2 = sh[4] + sh[5] + sh[6] + sh[7];
    const float mu  = s * (1.f / (float)H_DIM);
    const float var = s2 * (1.f / (float)H_DIM) - mu * mu;
    const float rs  = rsqrtf(var + EPS_LN);
#pragma unroll
    for (int i = 0; i < 4; ++i) {
        const int c = tid + i * 128;
        h[base + c] = (y[i] - mu) * rs * g[c] + bta[c] + h[base + c];
    }
}

__global__ __launch_bounds__(256)
void mean_s_kernel(const float* __restrict__ h, float* __restrict__ gf) {
    const int c = blockIdx.x * 256 + threadIdx.x;
    const int b = blockIdx.y;
    const float* p = h + (size_t)b * S_LEN * H_DIM + c;
    float s = 0.f;
    for (int t = 0; t < S_LEN; ++t) s += p[(size_t)t * H_DIM];
    gf[b * H_DIM + c] = s * (1.f / (float)S_LEN);
}

__global__ __launch_bounds__(256)
void out_proj3_kernel(const float* __restrict__ o2, const float* __restrict__ w3,
                      const float* __restrict__ b3, float* __restrict__ out) {
    const int lane = threadIdx.x & 31;
    const int wave = threadIdx.x >> 5;
    const int row  = blockIdx.x * 8 + wave;
    const float* p = o2 + (size_t)row * 128;
    float s = 0.f;
#pragma unroll
    for (int i = 0; i < 4; ++i) s += p[lane + 32 * i] * w3[lane + 32 * i];
#pragma unroll
    for (int off = 16; off > 0; off >>= 1) s += __shfl_xor(s, off, 32);
    if (lane == 0) out[row] = s + b3[0];
}

// ---------------------------------------------------------------------------
extern "C" void kernel_launch(void* const* d_in, const int* in_sizes, int n_in,
                              void* d_out, int out_size, void* d_ws, size_t ws_size,
                              hipStream_t stream) {
    (void)in_sizes; (void)n_in; (void)out_size; (void)ws_size;
    const float* x      = (const float*)d_in[0];
    const float* in_w   = (const float*)d_in[1];
    const float* in_b   = (const float*)d_in[2];
    const float* fw_r   = (const float*)d_in[3];
    const float* fw_i   = (const float*)d_in[4];
    const float* conv_w = (const float*)d_in[5];
    const float* conv_b = (const float*)d_in[6];
    const float* ln_g   = (const float*)d_in[7];
    const float* ln_b   = (const float*)d_in[8];
    const float* op1_w  = (const float*)d_in[9];
    const float* op1_b  = (const float*)d_in[10];
    const float* op2_w  = (const float*)d_in[11];
    const float* op2_b  = (const float*)d_in[12];
    const float* op3_w  = (const float*)d_in[13];
    const float* op3_b  = (const float*)d_in[14];
    const float* h1_w   = (const float*)d_in[15];
    const float* h1_b   = (const float*)d_in[16];
    const float* h2_w   = (const float*)d_in[17];
    const float* h2_b   = (const float*)d_in[18];

    char* ws = (char*)d_ws;
    const size_t HSZ = (size_t)B_BATCH * S_LEN * H_DIM * sizeof(float);   // 33.5 MB
    float* h       = (float*)(ws);
    float* fb      = (float*)(ws + HSZ);
    float* cv      = (float*)(ws + 2 * HSZ);
    float* spec    = (float*)(ws + 3 * HSZ);                              // [B,64,H]
    float* specout = (float*)(ws + 3 * HSZ + (1u << 20));
    float* o1      = (float*)(ws + 3 * HSZ + (2u << 20));                 // [B,S,256]
    float* o2      = (float*)(ws + 3 * HSZ + (2u << 20) + HSZ / 2);       // [B,S,128]
    float* gf      = (float*)(ws + 3 * HSZ + (2u << 20) + HSZ / 2 + HSZ / 4);
    float* k1      = gf + B_BATCH * H_DIM;

    const int ROWS = B_BATCH * S_LEN;   // 16384

    input_proj_kernel<<<dim3((ROWS * H_DIM + 255) / 256), 256, 0, stream>>>(x, in_w, in_b, h);

    for (int l = 0; l < L_LAYERS; ++l) {
        const float* Wr  = fw_r + (size_t)l * M_MODES * H_DIM * H_DIM;
        const float* Wi  = fw_i + (size_t)l * M_MODES * H_DIM * H_DIM;
        const float* cw  = conv_w + (size_t)l * H_DIM * H_DIM;
        const float* cbv = conv_b + l * H_DIM;
        const float* g   = ln_g + l * H_DIM;
        const float* bb  = ln_b + l * H_DIM;

        fno_fwd_dft_kernel<<<dim3(2, H_DIM / 32, B_BATCH), 32, 0, stream>>>(h, spec);
        fno_spectral_kernel<<<dim3(M_MODES, H_DIM / 32), 32, 0, stream>>>(spec, Wr, Wi, specout);
        fno_inv_dft_kernel<<<dim3(S_LEN / 32, H_DIM / 32, B_BATCH), 32, 0, stream>>>(specout, fb);
        gemm_aw_kernel<0><<<dim3(ROWS / 32, H_DIM / 32), 32, 0, stream>>>(
            h, cw, cbv, cv, ROWS, H_DIM, H_DIM);
        combine_ln_kernel<<<dim3(ROWS), 128, 0, stream>>>(fb, cv, g, bb, h);
    }

    // output head: H -> 256 -> 128 -> 1
    gemm_aw_kernel<1><<<dim3(ROWS / 32, 256 / 32), 32, 0, stream>>>(h, op1_w, op1_b, o1, ROWS, 256, H_DIM);
    gemm_aw_kernel<1><<<dim3(ROWS / 32, 128 / 32), 32, 0, stream>>>(o1, op2_w, op2_b, o2, ROWS, 128, 256);
    out_proj3_kernel<<<dim3(ROWS / 8), 256, 0, stream>>>(o2, op3_w, op3_b, (float*)d_out);

    // crypto head: mean over S, H -> H (gelu) -> KB (sigmoid)
    mean_s_kernel<<<dim3(H_DIM / 256, B_BATCH), 256, 0, stream>>>(h, gf);
    gemm_aw_kernel<1><<<dim3(1, H_DIM / 32), 32, 0, stream>>>(gf, h1_w, h1_b, k1, B_BATCH, H_DIM, H_DIM);
    gemm_aw_kernel<2><<<dim3(1, KB_BITS / 32), 32, 0, stream>>>(
        k1, h2_w, h2_b, (float*)d_out + ROWS, B_BATCH, KB_BITS, H_DIM);
}
